// SpectralConv1d_58798102282307
// MI455X (gfx1250) — matrix operand order repaired and checked by reading, not compiled, not run on hardware
//
#include <hip/hip_runtime.h>
#include <hip/hip_bf16.h>

typedef __attribute__((ext_vector_type(16))) _Float16 v16h;
typedef __attribute__((ext_vector_type(8)))  _Float16 v8h;
typedef __attribute__((ext_vector_type(8)))  float    v8f;
typedef __attribute__((ext_vector_type(4)))  float    v4f;
typedef __attribute__((ext_vector_type(4)))  int      v4i;

#define B_SZ   32
#define CIN    64
#define COUT   64
#define MODES  128
#define NLEN   16384
#define ROWS   (B_SZ * CIN)   /* 2048 */
#define K2     (2 * MODES)    /* 256  */
#define KSPLIT 16

#define BM 128
#define BN 128
#define BK 32
#define LDA 40
#define LDB 40

#define TWO_PI_OVER_N 3.834951969714103e-4f  /* 2*pi/16384 */

// -------- CDNA5 async global->LDS DMA (ASYNCcnt path), with safe fallback ---
#if defined(__has_builtin)
#  if __has_builtin(__builtin_amdgcn_global_load_async_to_lds_b128)
#    define HAVE_ASYNC_LDS 1
#  endif
#  if __has_builtin(__builtin_amdgcn_s_wait_asynccnt)
#    define HAVE_WAIT_ASYNC 1
#  endif
#endif
#ifndef HAVE_ASYNC_LDS
#  define HAVE_ASYNC_LDS 0
#endif
#ifndef HAVE_WAIT_ASYNC
#  define HAVE_WAIT_ASYNC 0
#endif

#define AS1 __attribute__((address_space(1)))
#define AS3 __attribute__((address_space(3)))

#if HAVE_ASYNC_LDS
__device__ __forceinline__ void async_cp_b128(_Float16* lds, const _Float16* g) {
  // signature: (v4i AS1* gaddr, v4i AS3* lds, imm offset, imm cpol)
  __builtin_amdgcn_global_load_async_to_lds_b128((AS1 v4i*)g, (AS3 v4i*)lds,
                                                 0, 0);
}
#endif

__device__ __forceinline__ void cp_b128_sync(_Float16* lds, const _Float16* g) {
  *(v8h*)lds = *(const v8h*)g;
}

__device__ __forceinline__ void copy16h(_Float16* lds, const _Float16* g) {
#if HAVE_ASYNC_LDS
  async_cp_b128(lds, g);
  async_cp_b128(lds + 8, g + 8);
#else
  cp_b128_sync(lds, g);
  cp_b128_sync(lds + 8, g + 8);
#endif
}

__device__ __forceinline__ void wait_async_then_barrier() {
#if HAVE_ASYNC_LDS
#  if HAVE_WAIT_ASYNC
  __builtin_amdgcn_s_wait_asynccnt(0);
#  else
  asm volatile("s_wait_asynccnt 0" ::: "memory");
#  endif
#endif
  __syncthreads();
}

// ---------------- twiddle generation (f16, transposed for B-operand use) ----
// Bt1[kk][n] = cos(2*pi*(kk*n mod N)/N), kk<128, n<16384   (stage-1 B, K dim = n)
__global__ __launch_bounds__(256) void gen_tw1(_Float16* __restrict__ Bt1) {
  int idx = blockIdx.x * 256 + threadIdx.x;          // 128*16384 total
  int kk = idx >> 14;
  int n  = idx & (NLEN - 1);
  int r  = (kk * n) & (NLEN - 1);
  Bt1[idx] = (_Float16)__cosf((float)r * TWO_PI_OVER_N);
}

// Bt3[n][mp]: mp<128 -> cos(2*pi*m*n/N); mp>=128 -> -sin(...)  (stage-3 B, K dim = mp)
__global__ __launch_bounds__(256) void gen_tw3(_Float16* __restrict__ Bt3) {
  int idx = blockIdx.x * 256 + threadIdx.x;          // 16384*256 total
  int mp = idx & (K2 - 1);
  int n  = idx >> 8;
  int m  = mp & (MODES - 1);
  int r  = (m * n) & (NLEN - 1);
  float s, c;
  __sincosf((float)r * TWO_PI_OVER_N, &s, &c);
  Bt3[idx] = (_Float16)((mp < MODES) ? c : -s);
}

// ---------------- tile staging: 128 rows x 32 halves for A and B ------------
// Thread t handles row = t>>1, 16-half chunk = (t&1)*16 for both operands.
template <bool A_F32>
__device__ __forceinline__ void stage_tile(
    const void* __restrict__ Aptr, const _Float16* __restrict__ Bt,
    _Float16* __restrict__ AsBuf, _Float16* __restrict__ BsBuf,
    int m0, int n0, int kg, int lda_g, int ldb_g, int t)
{
  const int row  = t >> 1;
  const int part = (t & 1) * 16;
  // ---- A tile ----
  if (A_F32) {
    const float* ap = (const float*)Aptr + (size_t)(m0 + row) * lda_g + kg + part;
    v4f f0 = *(const v4f*)ap;
    v4f f1 = *(const v4f*)(ap + 4);
    v4f f2 = *(const v4f*)(ap + 8);
    v4f f3 = *(const v4f*)(ap + 12);
    v8h h0, h1;
#pragma unroll
    for (int i = 0; i < 4; ++i) {
      h0[i] = (_Float16)f0[i]; h0[4 + i] = (_Float16)f1[i];
      h1[i] = (_Float16)f2[i]; h1[4 + i] = (_Float16)f3[i];
    }
    *(v8h*)&AsBuf[row * LDA + part]     = h0;
    *(v8h*)&AsBuf[row * LDA + part + 8] = h1;
  } else {
    const _Float16* ap = (const _Float16*)Aptr + (size_t)(m0 + row) * lda_g + kg + part;
    copy16h(&AsBuf[row * LDA + part], ap);
  }
  // ---- B tile (Bt is output-column-major: element (k, col) = Bt[col*ldb_g+k]) ----
  const _Float16* bp = Bt + (size_t)(n0 + row) * ldb_g + kg + part;
  copy16h(&BsBuf[row * LDB + part], bp);
}

// ---------------- WMMA GEMM: C = alpha * A @ Bt^T ---------------------------
// 256 threads = 8 wave32s arranged 4(m) x 2(n); wave tile = 32 rows x 64 cols.
// Double-buffered LDS; async DMA staging where available.
template <bool A_F32>
__global__ __launch_bounds__(256) void gemm_wmma(
    const void* __restrict__ Aptr, const _Float16* __restrict__ Bt,
    float* __restrict__ C, int lda_g, int ldb_g, int ldc,
    int k_len, float alpha, long long zstride)
{
  __shared__ _Float16 As[2][BM * LDA];
  __shared__ _Float16 Bs[2][BN * LDB];

  const int t   = threadIdx.x;
  const int m0  = blockIdx.x * BM;
  const int n0  = blockIdx.y * BN;
  const int kz0 = blockIdx.z * k_len;

  const int wave = t >> 5;
  const int lane = t & 31;
  const int mw   = wave >> 1;    // 0..3 : 32-row strip
  const int nw   = wave & 1;     // 0..1 : 64-col strip
  const int l15  = lane & 15;
  const int hi   = lane >> 4;

  v8f acc[2][4] = {};

  // Prologue: fill buffer 0
  stage_tile<A_F32>(Aptr, Bt, As[0], Bs[0], m0, n0, kz0, lda_g, ldb_g, t);

  const int iters = k_len / BK;
  for (int i = 0; i < iters; ++i) {
    const int p = i & 1;
    wait_async_then_barrier();     // buffer p complete for all waves

    if (i + 1 < iters)             // kick off DMA for the next tile
      stage_tile<A_F32>(Aptr, Bt, As[p ^ 1], Bs[p ^ 1], m0, n0,
                        kz0 + (i + 1) * BK, lda_g, ldb_g, t);

    const _Float16* Asb = As[p];
    const _Float16* Bsb = Bs[p];

    // Two A fragments (rows mw*32 + s*16): lane l15 = M; slots 0-7 = K(hi*8..),
    // slots 8-15 = K+16 (ISA 16-bit A layout).
    v16h a[2];
#pragma unroll
    for (int s = 0; s < 2; ++s) {
      const _Float16* arow = &Asb[(mw * 32 + s * 16 + l15) * LDA];
      v8h lo = *(const v8h*)(arow + hi * 8);
      v8h hh = *(const v8h*)(arow + 16 + hi * 8);
      a[s] = __builtin_shufflevector(lo, hh,
                                     0,1,2,3,4,5,6,7,8,9,10,11,12,13,14,15);
    }

#pragma unroll
    for (int jt = 0; jt < 4; ++jt) {
      // B fragment: lane l15 = col N; slots 0-7 = K(hi*8..), slots 8-15 = K+16.
      const _Float16* brow = &Bsb[(nw * 64 + jt * 16 + l15) * LDB];
      v8h b_lo = *(const v8h*)(brow + hi * 8);
      v8h b_hi = *(const v8h*)(brow + 16 + hi * 8);
      v16h b = __builtin_shufflevector(b_lo, b_hi,
                                       0,1,2,3,4,5,6,7,8,9,10,11,12,13,14,15);
      acc[0][jt] = __builtin_amdgcn_wmma_f32_16x16x32_f16(
          false, a[0], false, b, (short)0, acc[0][jt], false, false);
      acc[1][jt] = __builtin_amdgcn_wmma_f32_16x16x32_f16(
          false, a[1], false, b, (short)0, acc[1][jt], false, false);
    }
  }

  // D layout: VGPR r -> M = r + 8*hi ; N = l15
  float* Cz = C + (long long)blockIdx.z * zstride;
#pragma unroll
  for (int s = 0; s < 2; ++s) {
#pragma unroll
    for (int jt = 0; jt < 4; ++jt) {
      const int col = n0 + nw * 64 + jt * 16 + l15;
#pragma unroll
      for (int r = 0; r < 8; ++r) {
        const int row = m0 + mw * 32 + s * 16 + hi * 8 + r;
        Cz[(size_t)row * ldc + col] = alpha * acc[s][jt][r];
      }
    }
  }
}

// ---------------- K-split reduction of stage-1 partials ---------------------
__global__ __launch_bounds__(256) void reduce_k(const float* __restrict__ Xrp,
                                                float* __restrict__ Xr) {
  int idx = blockIdx.x * 256 + threadIdx.x;   // ROWS*MODES
  float s = 0.f;
#pragma unroll
  for (int z = 0; z < KSPLIT; ++z) s += Xrp[(size_t)z * ROWS * MODES + idx];
  Xr[idx] = s;
}

// ---------------- stage 2: spectral mixing, pack f16 A2 = [out_r | out_i] ---
__global__ __launch_bounds__(256) void stage2(const float* __restrict__ Xr,
                                              const float* __restrict__ wr,
                                              const float* __restrict__ wi,
                                              _Float16* __restrict__ A2) {
  int idx = blockIdx.x * 256 + threadIdx.x;   // B*COUT*MODES = 262144
  int m = idx & (MODES - 1);
  int o = (idx >> 7) & (COUT - 1);
  int b = idx >> 13;
  float sr = 0.f, si = 0.f;
  for (int i = 0; i < CIN; ++i) {
    float xv  = Xr[(size_t)(b * CIN + i) * MODES + m];
    sr += xv * wr[(size_t)(i * COUT + o) * MODES + m];
    si += xv * wi[(size_t)(i * COUT + o) * MODES + m];
  }
  int row = b * COUT + o;
  A2[(size_t)row * K2 + m]         = (_Float16)sr;
  A2[(size_t)row * K2 + MODES + m] = (_Float16)si;   // pairs with -sin rows of Bt3
}

// ---------------------------------------------------------------------------
extern "C" void kernel_launch(void* const* d_in, const int* in_sizes, int n_in,
                              void* d_out, int out_size, void* d_ws, size_t ws_size,
                              hipStream_t stream) {
  (void)in_sizes; (void)n_in; (void)out_size; (void)ws_size;
  const float* x  = (const float*)d_in[0];   // [32,64,16384]
  const float* wr = (const float*)d_in[1];   // [64,64,128]
  const float* wi = (const float*)d_in[2];   // [64,64,128]
  float* y = (float*)d_out;                  // [32,64,16384]

  char* ws = (char*)d_ws;
  _Float16* Bt1 = (_Float16*)(ws);                 //  4 MB : [128][16384] f16
  _Float16* Bt3 = (_Float16*)(ws + (4ull  << 20)); //  8 MB : [16384][256] f16
  float*    Xrp = (float*)   (ws + (12ull << 20)); // 16 MB : [16][2048][128] f32
  float*    Xr  = (float*)   (ws + (28ull << 20)); //  1 MB : [2048][128] f32
  _Float16* A2  = (_Float16*)(ws + (29ull << 20)); //  1 MB : [2048][256] f16

  gen_tw1<<<(MODES * NLEN) / 256, 256, 0, stream>>>(Bt1);
  gen_tw3<<<(NLEN * K2)   / 256, 256, 0, stream>>>(Bt3);

  // Stage 1: Xrp[z] = x @ cos^T  (M=2048, N=128, K=16384 split 16 ways)
  gemm_wmma<true><<<dim3(ROWS / BM, 1, KSPLIT), 256, 0, stream>>>(
      x, Bt1, Xrp, NLEN, NLEN, MODES, NLEN / KSPLIT, 1.0f,
      (long long)ROWS * MODES);

  reduce_k<<<(ROWS * MODES) / 256, 256, 0, stream>>>(Xrp, Xr);
  stage2<<<(B_SZ * COUT * MODES) / 256, 256, 0, stream>>>(Xr, wr, wi, A2);

  // Stage 3: y = (A2 @ [cos;-sin]) / N  (M=2048, N=16384, K=256)
  gemm_wmma<false><<<dim3(ROWS / BM, NLEN / BN, 1), 256, 0, stream>>>(
      A2, Bt3, y, K2, K2, NLEN, K2, 1.0f / (float)NLEN, 0ll);
}
